// IntraCL_36670430773285
// MI455X (gfx1250) — compile-verified, matrix operand order checked by reading, not run on hardware
//
#include <hip/hip_runtime.h>
#include <hip/hip_bf16.h>
#include <math.h>

typedef __attribute__((ext_vector_type(16))) _Float16 v16h;
typedef __attribute__((ext_vector_type(8)))  _Float16 v8h;
typedef __attribute__((ext_vector_type(4)))  _Float16 v4h;
typedef __attribute__((ext_vector_type(8)))  float    v8f;

#define BS 8192
#define DD 128
#define NCHUNK 8                 // column-sweep split per row tile (parallelism)
#define TILES_PER_CHUNK ((BS / 16) / NCHUNK)   // 64
#define NLOSSBLK 32              // loss kernel blocks (8192/256)

// ---------------------------------------------------------------------------
// Kernel 0: zero the class-1 counter (everything else is written-once).
// ---------------------------------------------------------------------------
__global__ void init_kernel(int* count) {
    if (threadIdx.x == 0) count[0] = 0;
}

// ---------------------------------------------------------------------------
// Kernel 1: L2-normalize rows (F.normalize eps semantics), convert to f16,
//           and count class-1 labels. One wave32 per row, 4 floats per lane.
// ---------------------------------------------------------------------------
__global__ __launch_bounds__(256) void norm_kernel(const float* __restrict__ data,
                                                   const int*   __restrict__ label,
                                                   _Float16*    __restrict__ x,
                                                   int*         __restrict__ count) {
    const int row  = blockIdx.x * 8 + (threadIdx.x >> 5);
    const int lane = threadIdx.x & 31;
    const float4 v = ((const float4*)(data + (size_t)row * DD))[lane];
    float ss = v.x * v.x + v.y * v.y + v.z * v.z + v.w * v.w;
    #pragma unroll
    for (int off = 16; off >= 1; off >>= 1) ss += __shfl_xor(ss, off, 32);
    const float inv = 1.0f / fmaxf(sqrtf(ss), 1e-12f);
    v4h h;
    h[0] = (_Float16)(v.x * inv);
    h[1] = (_Float16)(v.y * inv);
    h[2] = (_Float16)(v.z * inv);
    h[3] = (_Float16)(v.w * inv);
    *(v4h*)(x + (size_t)row * DD + lane * 4) = h;
    if (lane == 0) atomicAdd(count, (label[row] != 0) ? 1 : 0);
}

// ---------------------------------------------------------------------------
// Fragment loader for 16-bit WMMA A/B operands.
// 16x32 (MxK) layout: lane m = L%16 holds row m; K-chunks split by lane half:
//   lanes 0-15:  K = [k0..k0+7]  and [k0+16..k0+23]
//   lanes 16-31: K = [k0+8..k0+15] and [k0+24..k0+31]
// For x·xT, B columns are rows of x again, so A and B load identically.
// ---------------------------------------------------------------------------
__device__ inline v16h load_frag(const _Float16* __restrict__ rowp, int k0, int hi) {
    const v8h lo = *(const v8h*)(rowp + k0 + hi * 8);
    const v8h up = *(const v8h*)(rowp + k0 + 16 + hi * 8);
    return __builtin_shufflevector(lo, up, 0, 1, 2, 3, 4, 5, 6, 7,
                                           8, 9, 10, 11, 12, 13, 14, 15);
}

// ---------------------------------------------------------------------------
// Kernel 2: fused WMMA GEMM (x·xT, scaled) + per-row statistics.
// Each wave owns (row-tile, column-chunk): 16 rows x 1024 columns (64 tiles).
// A fragments (K=128) stay in registers for the whole sweep. Static softmax
// max m = e^scale is exact (row max is the diagonal, cosines <= 1).
// C-fragment layout: VGPR r, lane L -> row = r + (L>=16)*8, col = L%16.
// Partial stats layout: P[(chunk*5 + stat)*BS + row], written exactly once.
// ---------------------------------------------------------------------------
__global__ __launch_bounds__(128) void stats_kernel(const _Float16* __restrict__ x,
                                                    const int*      __restrict__ label,
                                                    const float*    __restrict__ scale,
                                                    float*          __restrict__ P) {
    const int lane    = threadIdx.x & 31;
    const int wave    = threadIdx.x >> 5;
    const int wglob   = blockIdx.x * 4 + wave;
    const int rowbase = (wglob >> 3) * 16;          // 512 row tiles
    const int chunk   = wglob & (NCHUNK - 1);       // 8 column chunks
    const int m       = lane & 15;
    const int hi      = lane >> 4;
    const float escale = __expf(scale[0]);

    const _Float16* arow = x + (size_t)(rowbase + m) * DD;
    const v16h a0 = load_frag(arow, 0,  hi);
    const v16h a1 = load_frag(arow, 32, hi);
    const v16h a2 = load_frag(arow, 64, hi);
    const v16h a3 = load_frag(arow, 96, hi);

    int labrow[8];
    #pragma unroll
    for (int r = 0; r < 8; ++r) labrow[r] = label[rowbase + r + hi * 8];

    float accS[8], accZ[8], accW[8], accSs[8], accZs[8];
    #pragma unroll
    for (int r = 0; r < 8; ++r) {
        accS[r] = 0.f; accZ[r] = 0.f; accW[r] = 0.f; accSs[r] = 0.f; accZs[r] = 0.f;
    }

    const int ct0 = chunk * TILES_PER_CHUNK;
    for (int t = 0; t < TILES_PER_CHUNK; ++t) {
        const int colbase = (ct0 + t) * 16;
        const _Float16* brow = x + (size_t)(colbase + m) * DD;
        const v16h b0 = load_frag(brow, 0,  hi);
        const v16h b1 = load_frag(brow, 32, hi);
        const v16h b2 = load_frag(brow, 64, hi);
        const v16h b3 = load_frag(brow, 96, hi);
        const int labc = label[colbase + m];

        v8f c = {};
        c = __builtin_amdgcn_wmma_f32_16x16x32_f16(false, a0, false, b0, (short)0, c, false, false);
        c = __builtin_amdgcn_wmma_f32_16x16x32_f16(false, a1, false, b1, (short)0, c, false, false);
        c = __builtin_amdgcn_wmma_f32_16x16x32_f16(false, a2, false, b2, (short)0, c, false, false);
        c = __builtin_amdgcn_wmma_f32_16x16x32_f16(false, a3, false, b3, (short)0, c, false, false);

        #pragma unroll
        for (int r = 0; r < 8; ++r) {
            const float logit = c[r] * escale;               // cosine * e^scale
            const float e     = __expf(logit - escale);      // max-subtracted, arg <= 0
            const float fs    = (labc == labrow[r]) ? 1.0f : 0.0f;
            accS[r]  += logit;
            accZ[r]  += e;
            accW[r]   = fmaf(e,  logit, accW[r]);
            accSs[r]  = fmaf(fs, logit, accSs[r]);
            accZs[r]  = fmaf(fs, e,     accZs[r]);
        }
    }

    // Butterfly-reduce over the 16 lanes that share each row (masks 1..8 stay
    // inside a 16-lane half of the wave32).
    #pragma unroll
    for (int r = 0; r < 8; ++r) {
        #pragma unroll
        for (int off = 1; off < 16; off <<= 1) {
            accS[r]  += __shfl_xor(accS[r],  off, 32);
            accZ[r]  += __shfl_xor(accZ[r],  off, 32);
            accW[r]  += __shfl_xor(accW[r],  off, 32);
            accSs[r] += __shfl_xor(accSs[r], off, 32);
            accZs[r] += __shfl_xor(accZs[r], off, 32);
        }
    }
    if (m == 0) {  // lane 0 -> rows 0..7, lane 16 -> rows 8..15
        float* Pc = P + (size_t)chunk * 5 * BS;
        #pragma unroll
        for (int r = 0; r < 8; ++r) {
            const int row = rowbase + r + hi * 8;
            Pc[0 * BS + row] = accS[r];
            Pc[1 * BS + row] = accZ[r];
            Pc[2 * BS + row] = accW[r];
            Pc[3 * BS + row] = accSs[r];
            Pc[4 * BS + row] = accZs[r];
        }
    }
}

// ---------------------------------------------------------------------------
// Kernel 3: combine chunk partials, closed-form per-row loss, block-reduce.
//   q per row has 2 values: qs = e^{1/c}/Zq (same class), qd = 1/Zq, with
//   Zq = c·e^{1/c} + (bs-c).  L_i = m + log Z_i.
//   loss1_i = H - qd·S - (qs-qd)·Ssame + L       (H = Σ q log q, closed form)
//   loss2_i = (W/Z - L) - (lqd + (lqs-lqd)·Zs/Z)
// Deterministic: fixed-order chunk sum, fixed-shape block reduction.
// ---------------------------------------------------------------------------
__global__ __launch_bounds__(256) void loss_kernel(const float* __restrict__ P,
                                                   const int*   __restrict__ label,
                                                   const int*   __restrict__ count,
                                                   const float* __restrict__ scale,
                                                   float*       __restrict__ blockpart) {
    const int i = blockIdx.x * blockDim.x + threadIdx.x;

    float Si = 0.f, Zi = 0.f, Wi = 0.f, Ssi = 0.f, Zsi = 0.f;
    #pragma unroll
    for (int ch = 0; ch < NCHUNK; ++ch) {
        const float* Pc = P + (size_t)ch * 5 * BS;
        Si  += Pc[0 * BS + i];
        Zi  += Pc[1 * BS + i];
        Wi  += Pc[2 * BS + i];
        Ssi += Pc[3 * BS + i];
        Zsi += Pc[4 * BS + i];
    }

    const float escale = __expf(scale[0]);
    const int N1 = count[0];
    const int c  = (label[i] != 0) ? N1 : (BS - N1);
    const float cf = (float)c, df = (float)(BS - c);

    const float einv = __expf(1.0f / cf);
    const float Zq   = cf * einv + df;
    const float lZq  = __logf(Zq);
    const float qs   = einv / Zq;
    const float qd   = 1.0f / Zq;
    const float lqs  = 1.0f / cf - lZq;
    const float lqd  = -lZq;
    const float H    = cf * qs * lqs + df * qd * lqd;

    const float L  = escale + __logf(Zi);
    const float loss1 = H - qd * Si - (qs - qd) * Ssi + L;
    const float loss2 = (Wi / Zi - L) - (lqd + (lqs - lqd) * Zsi / Zi);

    float v = (loss1 + loss2) * (0.5f / (float)BS);
    #pragma unroll
    for (int off = 16; off >= 1; off >>= 1) v += __shfl_xor(v, off, 32);

    __shared__ float sdata[8];
    const int wv = threadIdx.x >> 5;
    if ((threadIdx.x & 31) == 0) sdata[wv] = v;
    __syncthreads();
    if (threadIdx.x == 0) {
        float s = 0.f;
        #pragma unroll
        for (int w = 0; w < 8; ++w) s += sdata[w];
        blockpart[blockIdx.x] = s;
    }
}

// ---------------------------------------------------------------------------
// Kernel 4: final deterministic sum of the 32 block partials.
// ---------------------------------------------------------------------------
__global__ void final_kernel(const float* __restrict__ blockpart, float* __restrict__ out) {
    if (threadIdx.x == 0) {
        float s = 0.f;
        for (int b = 0; b < NLOSSBLK; ++b) s += blockpart[b];
        out[0] = s;
    }
}

// ---------------------------------------------------------------------------
extern "C" void kernel_launch(void* const* d_in, const int* in_sizes, int n_in,
                              void* d_out, int out_size, void* d_ws, size_t ws_size,
                              hipStream_t stream) {
    const float* data  = (const float*)d_in[0];
    const float* scale = (const float*)d_in[1];
    const int*   label = (const int*)d_in[2];
    float* out = (float*)d_out;

    // Workspace layout
    _Float16* x = (_Float16*)d_ws;                              // 2 MB
    float* P = (float*)((char*)d_ws + (size_t)BS * DD * sizeof(_Float16));
    //   P: NCHUNK * 5 * BS floats = 1.25 MB (each slot written exactly once)
    float* blockpart = P + (size_t)NCHUNK * 5 * BS;             // 32 floats
    int*   count = (int*)(blockpart + NLOSSBLK);

    init_kernel<<<1, 32, 0, stream>>>(count);
    norm_kernel<<<BS / 8, 256, 0, stream>>>(data, label, x, count);
    // 512 row tiles * 8 column chunks = 4096 waves, 4 waves per block
    stats_kernel<<<(BS / 16) * NCHUNK / 4, 128, 0, stream>>>(x, label, scale, P);
    loss_kernel<<<NLOSSBLK, 256, 0, stream>>>(P, label, count, scale, blockpart);
    final_kernel<<<1, 32, 0, stream>>>(blockpart, out);
}
